// NodeTaskHead_25194278158877
// MI455X (gfx1250) — matrix operand order repaired and checked by reading, not compiled, not run on hardware
//
#include <hip/hip_runtime.h>
#include <hip/hip_bf16.h>

// ---------------- problem constants ----------------
#define B_    8
#define N_    256
#define M_    512
#define E_    1024
#define H_    32
#define D_    32
#define EXT_  768            // N + M

static constexpr float SCALING = 0.17677669529663687f;   // 32^-0.5
static constexpr float NEG_BIG = -3.4028234663852886e38f; // finfo(f32).min

// ---------------- vector types for WMMA ----------------
typedef __attribute__((ext_vector_type(16))) __bf16   v16bf;
typedef __attribute__((ext_vector_type(8)))  float    v8f;
typedef __attribute__((ext_vector_type(8)))  unsigned v8u;
typedef __attribute__((ext_vector_type(4)))  float    v4f;

// hardware bf16 conversion (native cvt on gfx1250, RNE)
__device__ __forceinline__ unsigned short f2bf(float f) {
  return __builtin_bit_cast(unsigned short, (__bf16)f);
}
__device__ __forceinline__ unsigned pack2(float lo, float hi) {
  return (unsigned)f2bf(lo) | ((unsigned)f2bf(hi) << 16);
}
__device__ __forceinline__ v8f wmma_bf16(v16bf a, v16bf b, v8f c) {
  return __builtin_amdgcn_wmma_f32_16x16x32_bf16(false, a, false, b, (short)0, c,
                                                 false, false);
}
// 8 consecutive dwords (B fragments / contiguous 32 bf16)
__device__ __forceinline__ v8u load8u(const unsigned* p) {
  uint4 l0 = *(const uint4*)(p);
  uint4 l1 = *(const uint4*)(p + 4);
  v8u r;
  r[0] = l0.x; r[1] = l0.y; r[2] = l0.z; r[3] = l0.w;
  r[4] = l1.x; r[5] = l1.y; r[6] = l1.z; r[7] = l1.w;
  return r;
}
// dwords {0..3, 8..11} (A-fragment split runs)
__device__ __forceinline__ v8u load8u_split(const unsigned* p) {
  uint4 l0 = *(const uint4*)(p);
  uint4 l1 = *(const uint4*)(p + 8);
  v8u r;
  r[0] = l0.x; r[1] = l0.y; r[2] = l0.z; r[3] = l0.w;
  r[4] = l1.x; r[5] = l1.y; r[6] = l1.z; r[7] = l1.w;
  return r;
}

// =====================================================================
// Kernel 0: one-shot f32 -> bf16 conversion (x and the 6 weight mats)
// =====================================================================
__global__ __launch_bounds__(256)
void cvt_bf16_kernel(const float* __restrict__ in,
                     unsigned short* __restrict__ out, int n4) {
  int i = blockIdx.x * 256 + threadIdx.x;
  if (i < n4) {
    v4f f = ((const v4f*)in)[i];
    uint2 o;
    o.x = pack2(f.x, f.y);
    o.y = pack2(f.z, f.w);
    ((uint2*)out)[i] = o;
  }
}

// =====================================================================
// Kernel 1: projection GEMM (bf16 operands, f32 accum).
//   Ping-pong double buffered K-loop (no register-copy rotation).
//   out[row,:] = (x_row @ W^T) * scale  (bf16 result)
//   grid: (rows/128, E/64), block 128 (4 waves); wave = 32 rows x 64 cols.
// =====================================================================
__global__ __launch_bounds__(128)
void proj_kernel(const unsigned short* __restrict__ xbf,
                 const unsigned short* __restrict__ Wbf,
                 unsigned short* __restrict__ out, float scale, int out_ext) {
  const int tid  = threadIdx.x;
  const int wave = tid >> 5, lane = tid & 31;
  const int l15  = lane & 15;
  const int row0 = blockIdx.x * 128 + wave * 32;
  const int col0 = blockIdx.y * 64;

  const unsigned* arow[2];
#pragma unroll
  for (int ra = 0; ra < 2; ++ra) {
    int lrow = row0 + ra * 16 + l15;
    int b = lrow / N_, n = lrow % N_;
    arow[ra] = (const unsigned*)(xbf + ((size_t)n * B_ + b) * E_);
  }
  const unsigned* brow[4];
#pragma unroll
  for (int t = 0; t < 4; ++t)
    brow[t] = (const unsigned*)(Wbf + (size_t)(col0 + t * 16 + l15) * E_);

  const int aoff = (lane < 16) ? 0 : 4;
  const int boff = (lane < 16) ? 0 : 4;

  v8f acc[2][4] = {};
  v8u A0[2], B0[4], A1[2], B1[4];

  auto loadset = [&](v8u* A, v8u* Bf, int k) {
    const int kh = k >> 1;
    A[0] = load8u_split(arow[0] + kh + aoff);
    A[1] = load8u_split(arow[1] + kh + aoff);
#pragma unroll
    for (int t = 0; t < 4; ++t) Bf[t] = load8u_split(brow[t] + kh + boff);
  };
  auto mmset = [&](const v8u* A, const v8u* Bf) {
    const v16bf x0 = __builtin_bit_cast(v16bf, A[0]);
    const v16bf x1 = __builtin_bit_cast(v16bf, A[1]);
#pragma unroll
    for (int t = 0; t < 4; ++t) {
      const v16bf bb = __builtin_bit_cast(v16bf, Bf[t]);
      acc[0][t] = wmma_bf16(x0, bb, acc[0][t]);
      acc[1][t] = wmma_bf16(x1, bb, acc[1][t]);
    }
  };

  loadset(A0, B0, 0);
  for (int k0 = 0; k0 < E_; k0 += 64) {
    __builtin_prefetch(arow[0] + (k0 >> 1) + 64, 0, 1);
    loadset(A1, B1, k0 + 32);
    mmset(A0, B0);
    const int kn = (k0 + 64 < E_) ? (k0 + 64) : 0;   // branch-free wrap
    loadset(A0, B0, kn);
    mmset(A1, B1);
  }

  // C: VGPR g -> M = g (lanes 0-15) / 8+g (lanes 16-31)
  const int rbase = (lane < 16) ? 0 : 8;
#pragma unroll
  for (int ra = 0; ra < 2; ++ra) {
#pragma unroll
    for (int t = 0; t < 4; ++t) {
      const int col = col0 + t * 16 + l15;
#pragma unroll
      for (int g = 0; g < 8; ++g) {
        int rw = row0 + ra * 16 + rbase + g;
        int b2 = rw / N_, n2 = rw % N_;
        size_t drow = out_ext ? ((size_t)b2 * EXT_ + n2) : (size_t)rw;
        out[drow * E_ + col] = f2bf(acc[ra][t][g] * scale);
      }
    }
  }
}

// =====================================================================
// Kernel 2: gather expanded rows of k/v/ve:  row[b,N+m] = row[b, idx[b,m]]
// =====================================================================
__global__ __launch_bounds__(128)
void gather_kernel(const int* __restrict__ oidx, unsigned short* __restrict__ kbuf,
                   unsigned short* __restrict__ vbuf,
                   unsigned short* __restrict__ vebuf) {
  const int bm = blockIdx.x;
  const int b = bm / M_, m = bm % M_;
  const int src = oidx[bm];
  const size_t so = ((size_t)b * EXT_ + src) * E_;
  const size_t dof = ((size_t)b * EXT_ + N_ + m) * E_;
  const int t = threadIdx.x;                        // 128 threads, 16B each
  ((uint4*)(kbuf + dof))[t]  = ((const uint4*)(kbuf + so))[t];
  ((uint4*)(vbuf + dof))[t]  = ((const uint4*)(vbuf + so))[t];
  ((uint4*)(vebuf + dof))[t] = ((const uint4*)(vebuf + so))[t];
}

// =====================================================================
// Kernel 3: fused attention per (b, h, 16-row tile of N).
//   scores (WMMA, K=D=32) + bias + masks -> softmax in LDS ->
//   wave0: xo = P @ ve_h ; wave1..3: vec_c = (P * delta_c) @ v_h
//   P@V B-operands staged through LDS (transposed) per 32-K step.
// =====================================================================
__global__ __launch_bounds__(128)
void attn_kernel(const unsigned short* __restrict__ qb,
                 const unsigned short* __restrict__ kb,
                 const unsigned short* __restrict__ vb,
                 const unsigned short* __restrict__ veb,
                 const float* __restrict__ bias, const float* __restrict__ pos,
                 const float* __restrict__ epos,
                 const unsigned char* __restrict__ pmask,
                 const unsigned char* __restrict__ emask,
                 unsigned short* __restrict__ xoh,
                 unsigned short* __restrict__ vech) {
  __shared__ float S[16][EXT_ + 8];     // scores -> probs
  __shared__ float DN[16][EXT_ + 8];    // padding-zeroed 1/(dist+1)
  __shared__ float AP[EXT_][3];         // all_pos for this batch
  __shared__ float PS[16][3];           // pos of this row tile
  __shared__ float red[16][8];
  __shared__ float rmax[16], rsum[16];
  __shared__ unsigned short VT[2][32][36]; // [ve|v][col][k] 32x32 tiles, skewed

  const int id = blockIdx.x;
  const int nt = id & 15, h = (id >> 4) & (H_ - 1), b = id >> 9;
  const int tid = threadIdx.x, wave = tid >> 5, lane = tid & 31;
  const int n0 = nt * 16;

  // -------- phase A: geometry caches --------
  for (int m = tid; m < EXT_; m += 128) {
    const float* p = (m < N_) ? (pos + ((size_t)b * N_ + m) * 3)
                              : (epos + ((size_t)b * M_ + (m - N_)) * 3);
    AP[m][0] = p[0]; AP[m][1] = p[1]; AP[m][2] = p[2];
  }
  if (tid < 48) {
    int r = tid / 3, c = tid % 3;
    PS[r][c] = pos[((size_t)b * N_ + n0 + r) * 3 + c];
  }
  __syncthreads();
  for (int t = tid; t < 16 * EXT_; t += 128) {
    int r = t / EXT_, m = t - r * EXT_;
    int n = n0 + r;
    float dx = PS[r][0] - AP[m][0];
    float dy = PS[r][1] - AP[m][1];
    float dz = PS[r][2] - AP[m][2];
    float d = sqrtf(dx * dx + dy * dy + dz * dz);
    bool pm = pmask[b * N_ + n] != 0;
    bool fm = (m < N_) ? (pmask[b * N_ + m] != 0) : (emask[b * M_ + m - N_] != 0);
    if (pm || fm) d = 1.0e6f;
    DN[r][m] = pm ? 0.0f : 1.0f / (d + 1.0f);
  }

  // -------- phase B: scores via WMMA --------
  v8u qa;
  {
    const unsigned* qrow =
        (const unsigned*)(qb + ((size_t)b * N_ + n0 + (lane & 15)) * E_);
    qa = load8u_split(qrow + h * 16 + ((lane < 16) ? 0 : 4));
  }
  const v16bf qf = __builtin_bit_cast(v16bf, qa);

  for (int ct = wave; ct < EXT_ / 16; ct += 4) {
    const int mcol = ct * 16 + (lane & 15);
    const unsigned* krow = (const unsigned*)(kb + ((size_t)b * EXT_ + mcol) * E_);
    const int ub = h * 16 + ((lane < 16) ? 0 : 4);
    v8f acc = {};
    acc = wmma_bf16(qf, __builtin_bit_cast(v16bf, load8u_split(krow + ub)), acc);

    const int rbase = (lane < 16) ? 0 : 8;
#pragma unroll
    for (int g = 0; g < 8; ++g) {
      int r = rbase + g, n = n0 + r;
      float sv = acc[g] + bias[(((size_t)b * H_ + h) * N_ + n) * EXT_ + mcol];
      bool fm = (mcol < N_) ? (pmask[b * N_ + mcol] != 0)
                            : (emask[b * M_ + mcol - N_] != 0);
      bool pm = pmask[b * N_ + n] != 0;
      S[r][mcol] = (fm || pm) ? NEG_BIG : sv;
    }
  }
  __syncthreads();

  // -------- phase C: softmax over EXT (8 threads / row) --------
  {
    const int r = tid >> 3, g = tid & 7;
    float mx = NEG_BIG;
    for (int m = g; m < EXT_; m += 8) mx = fmaxf(mx, S[r][m]);
    red[r][g] = mx;
    __syncthreads();
    if (g == 0) {
      float m2 = red[r][0];
#pragma unroll
      for (int j = 1; j < 8; ++j) m2 = fmaxf(m2, red[r][j]);
      rmax[r] = m2;
    }
    __syncthreads();
    const float m2 = rmax[r];
    float s = 0.f;
    for (int m = g; m < EXT_; m += 8) {
      float e = __expf(S[r][m] - m2);
      S[r][m] = e; s += e;
    }
    red[r][g] = s;
    __syncthreads();
    if (g == 0) {
      float s2 = 0.f;
#pragma unroll
      for (int j = 0; j < 8; ++j) s2 += red[r][j];
      rsum[r] = 1.0f / s2;
    }
    __syncthreads();
    const float inv = rsum[r];
    for (int m = g; m < EXT_; m += 8) S[r][m] *= inv;
  }

  // -------- phase D: P @ V (wave0: xo via ve; wave1-3: vec_c via v) --------
  const int o = wave, c = o - 1;
  const int arow = lane & 15;
  const float pc0 = (o > 0) ? PS[arow][c] : 0.f;
  const int sV = (o == 0) ? 0 : 1;

  v8f acc0 = {}, acc1 = {};
  for (int kc = 0; kc < EXT_ / 32; ++kc) {
    const int k0 = kc * 32;
    __syncthreads();   // previous tile reads done / probs ready (first iter)
    // stage 32x32 bf16 V and VE tiles, transposed: VT[s][col][k]
    for (int idx = tid; idx < 32 * 16; idx += 128) {
      const int k = idx >> 4, cp = (idx & 15) * 2;
      const size_t roff = ((size_t)b * EXT_ + k0 + k) * E_ + h * D_ + cp;
      unsigned gve = *(const unsigned*)(veb + roff);
      unsigned gv  = *(const unsigned*)(vb + roff);
      VT[0][cp][k]     = (unsigned short)(gve & 0xFFFFu);
      VT[0][cp + 1][k] = (unsigned short)(gve >> 16);
      VT[1][cp][k]     = (unsigned short)(gv & 0xFFFFu);
      VT[1][cp + 1][k] = (unsigned short)(gv >> 16);
    }
    __syncthreads();

    // A frag from P (with delta_c folded in for the vec waves)
    const int kA = k0 + ((lane < 16) ? 0 : 8);
    v8u au;
#pragma unroll
    for (int j = 0; j < 4; ++j) {
      int m0 = kA + 2 * j, m2 = kA + 16 + 2 * j;
      float p0 = S[arow][m0], p1 = S[arow][m0 + 1];
      float p2 = S[arow][m2], p3 = S[arow][m2 + 1];
      if (o > 0) {
        p0 *= (pc0 - AP[m0][c]) * DN[arow][m0];
        p1 *= (pc0 - AP[m0 + 1][c]) * DN[arow][m0 + 1];
        p2 *= (pc0 - AP[m2][c]) * DN[arow][m2];
        p3 *= (pc0 - AP[m2 + 1][c]) * DN[arow][m2 + 1];
      }
      au[j] = pack2(p0, p1);
      au[4 + j] = pack2(p2, p3);
    }
    v16bf af = __builtin_bit_cast(v16bf, au);

    // B frags straight from the transposed LDS tile
#pragma unroll
    for (int dt = 0; dt < 2; ++dt) {
      const unsigned* tcol = (const unsigned*)&VT[sV][dt * 16 + (lane & 15)][0];
      const int kb8 = (lane < 16) ? 0 : 4;
      v8u bu;
#pragma unroll
      for (int j = 0; j < 8; ++j) bu[j] = tcol[kb8 + (j & 3) + (j >> 2) * 8];
      v8f& a = dt ? acc1 : acc0;
      a = wmma_bf16(af, __builtin_bit_cast(v16bf, bu), a);
    }
  }

  const int rbase = (lane < 16) ? 0 : 8;
#pragma unroll
  for (int dt = 0; dt < 2; ++dt) {
    v8f a = dt ? acc1 : acc0;
    const int col = h * D_ + dt * 16 + (lane & 15);
#pragma unroll
    for (int g = 0; g < 8; ++g) {
      int n = n0 + rbase + g;
      if (o == 0)
        xoh[((size_t)b * N_ + n) * E_ + col] = f2bf(a[g]);
      else
        vech[(((size_t)b * N_ + n) * 3 + c) * E_ + col] = f2bf(a[g]);
    }
  }
}

// =====================================================================
// Kernel 4: output GEMM (bf16 act, bf16 W, f32 out).
//   Ping-pong double buffered K-loop; grid: (rows/128, E/64), block 128.
// =====================================================================
__global__ __launch_bounds__(128)
void out_gemm_kernel(const unsigned short* __restrict__ act,
                     const unsigned short* __restrict__ Wbf,
                     float* __restrict__ out) {
  const int tid = threadIdx.x, wave = tid >> 5, lane = tid & 31;
  const int l15 = lane & 15;
  const int row0 = blockIdx.x * 128 + wave * 32;
  const int col0 = blockIdx.y * 64;

  const unsigned* arow[2];
#pragma unroll
  for (int ra = 0; ra < 2; ++ra)
    arow[ra] = (const unsigned*)(act + (size_t)(row0 + ra * 16 + l15) * E_);
  const unsigned* brow[4];
#pragma unroll
  for (int t = 0; t < 4; ++t)
    brow[t] = (const unsigned*)(Wbf + (size_t)(col0 + t * 16 + l15) * E_);

  const int aoff = (lane < 16) ? 0 : 4;
  const int boff = (lane < 16) ? 0 : 4;

  v8f acc[2][4] = {};
  v8u A0[2], B0[4], A1[2], B1[4];

  auto loadset = [&](v8u* A, v8u* Bf, int k) {
    const int kh = k >> 1;
    A[0] = load8u_split(arow[0] + kh + aoff);
    A[1] = load8u_split(arow[1] + kh + aoff);
#pragma unroll
    for (int t = 0; t < 4; ++t) Bf[t] = load8u_split(brow[t] + kh + boff);
  };
  auto mmset = [&](const v8u* A, const v8u* Bf) {
    const v16bf x0 = __builtin_bit_cast(v16bf, A[0]);
    const v16bf x1 = __builtin_bit_cast(v16bf, A[1]);
#pragma unroll
    for (int t = 0; t < 4; ++t) {
      const v16bf bb = __builtin_bit_cast(v16bf, Bf[t]);
      acc[0][t] = wmma_bf16(x0, bb, acc[0][t]);
      acc[1][t] = wmma_bf16(x1, bb, acc[1][t]);
    }
  };

  loadset(A0, B0, 0);
  for (int k0 = 0; k0 < E_; k0 += 64) {
    __builtin_prefetch(arow[0] + (k0 >> 1) + 64, 0, 1);
    loadset(A1, B1, k0 + 32);
    mmset(A0, B0);
    const int kn = (k0 + 64 < E_) ? (k0 + 64) : 0;
    loadset(A0, B0, kn);
    mmset(A1, B1);
  }

  const int rbase = (lane < 16) ? 0 : 8;
#pragma unroll
  for (int ra = 0; ra < 2; ++ra) {
#pragma unroll
    for (int t = 0; t < 4; ++t) {
      const int col = col0 + t * 16 + l15;
#pragma unroll
      for (int g = 0; g < 8; ++g)
        out[(size_t)(row0 + ra * 16 + rbase + g) * E_ + col] = acc[ra][t][g];
    }
  }
}

// =====================================================================
// host launcher
// =====================================================================
extern "C" void kernel_launch(void* const* d_in, const int* in_sizes, int n_in,
                              void* d_out, int out_size, void* d_ws,
                              size_t ws_size, hipStream_t stream) {
  const float* x    = (const float*)d_in[0];   // [N,B,E]
  const float* pos  = (const float*)d_in[1];   // [B,N,3]
  const float* epos = (const float*)d_in[2];   // [B,M,3]
  const float* bias = (const float*)d_in[3];   // [B,H,N,EXT]
  const unsigned char* pmask = (const unsigned char*)d_in[4];  // [B,N] bool
  const unsigned char* emask = (const unsigned char*)d_in[5];  // [B,M] bool
  const int* oidx = (const int*)d_in[6];       // [B,M]
  const float* Wq  = (const float*)d_in[7];
  const float* Wk  = (const float*)d_in[8];
  const float* Wv  = (const float*)d_in[9];
  const float* Wve = (const float*)d_in[10];
  const float* Wo  = (const float*)d_in[11];
  const float* Woe = (const float*)d_in[12];

  char* ws = (char*)d_ws;
  size_t off = 0;
  auto take = [&](size_t elems) {
    unsigned short* p = (unsigned short*)(ws + off);
    off += elems * 2;
    return p;
  };
  unsigned short* xbf   = take((size_t)N_ * B_ * E_);
  unsigned short* wqb   = take((size_t)E_ * E_);
  unsigned short* wkb   = take((size_t)E_ * E_);
  unsigned short* wvb   = take((size_t)E_ * E_);
  unsigned short* wveb  = take((size_t)E_ * E_);
  unsigned short* wob   = take((size_t)E_ * E_);
  unsigned short* woeb  = take((size_t)E_ * E_);
  unsigned short* qb    = take((size_t)B_ * N_ * E_);
  unsigned short* kbuf  = take((size_t)B_ * EXT_ * E_);
  unsigned short* vbuf  = take((size_t)B_ * EXT_ * E_);
  unsigned short* vebuf = take((size_t)B_ * EXT_ * E_);
  unsigned short* xoh   = take((size_t)B_ * N_ * E_);
  unsigned short* vech  = take((size_t)B_ * N_ * 3 * E_);
  (void)ws_size; (void)in_sizes; (void)n_in; (void)out_size;

  // --- one-shot bf16 conversions ---
  const int nx4 = (N_ * B_ * E_) / 4;        // x
  const int nw4 = (E_ * E_) / 4;             // each weight
  cvt_bf16_kernel<<<(nx4 + 255) / 256, 256, 0, stream>>>(x,   xbf,  nx4);
  cvt_bf16_kernel<<<(nw4 + 255) / 256, 256, 0, stream>>>(Wq,  wqb,  nw4);
  cvt_bf16_kernel<<<(nw4 + 255) / 256, 256, 0, stream>>>(Wk,  wkb,  nw4);
  cvt_bf16_kernel<<<(nw4 + 255) / 256, 256, 0, stream>>>(Wv,  wvb,  nw4);
  cvt_bf16_kernel<<<(nw4 + 255) / 256, 256, 0, stream>>>(Wve, wveb, nw4);
  cvt_bf16_kernel<<<(nw4 + 255) / 256, 256, 0, stream>>>(Wo,  wob,  nw4);
  cvt_bf16_kernel<<<(nw4 + 255) / 256, 256, 0, stream>>>(Woe, woeb, nw4);

  // --- projections (128-row blocks, 4 waves x 32 rows) ---
  const dim3 gproj((B_ * N_) / 128, E_ / 64);
  proj_kernel<<<gproj, 128, 0, stream>>>(xbf, wqb,  qb,    SCALING, 0);
  proj_kernel<<<gproj, 128, 0, stream>>>(xbf, wkb,  kbuf,  1.0f,    1);
  proj_kernel<<<gproj, 128, 0, stream>>>(xbf, wvb,  vbuf,  1.0f,    1);
  proj_kernel<<<gproj, 128, 0, stream>>>(xbf, wveb, vebuf, 1.0f,    1);

  gather_kernel<<<B_ * M_, 128, 0, stream>>>(oidx, kbuf, vbuf, vebuf);

  attn_kernel<<<B_ * H_ * (N_ / 16), 128, 0, stream>>>(
      qb, kbuf, vbuf, vebuf, bias, pos, epos, pmask, emask, xoh, vech);

  float* out_xo  = (float*)d_out;                        // [B,N,E]
  float* out_vec = (float*)d_out + (size_t)B_ * N_ * E_; // [B,N,3,E]
  out_gemm_kernel<<<dim3((B_ * N_) / 128, E_ / 64), 128, 0, stream>>>(xoh, woeb, out_xo);
  out_gemm_kernel<<<dim3((B_ * N_ * 3) / 128, E_ / 64), 128, 0, stream>>>(vech, wob, out_vec);
}
